// LNNModel_34540126995161
// MI455X (gfx1250) — matrix operand order repaired and checked by reading, not compiled, not run on hardware
//
#include <hip/hip_runtime.h>
#include <hip/hip_bf16.h>

typedef __bf16 bf16;
typedef __attribute__((ext_vector_type(16))) __bf16 v16bf;
typedef __attribute__((ext_vector_type(8)))  float  v8f;
typedef int v4i __attribute__((vector_size(16)));

#define BB 64
#define SS 512
#define FF 64
#define HH 512
#define LLAYERS 4
#define NHH 8
#define DHH 64
#define ALPHA_C 0.1f
#define EPS_LN 1e-5f
#define NWG_SCAN 8

// ---- CDNA5 async global->LDS support (guarded; falls back to sync copy) --
#if defined(__HIP_DEVICE_COMPILE__) && __has_builtin(__builtin_amdgcn_global_load_async_to_lds_b128)
#define HAVE_ASYNC_LDS 1
#else
#define HAVE_ASYNC_LDS 0
#endif

#if defined(__HIP_DEVICE_COMPILE__) && __has_builtin(__builtin_amdgcn_s_wait_asynccnt)
#define ASYNC_WAIT_0() __builtin_amdgcn_s_wait_asynccnt(0)
#define ASYNC_WAIT_2() __builtin_amdgcn_s_wait_asynccnt(2)
#elif HAVE_ASYNC_LDS
#define ASYNC_WAIT_0() asm volatile("s_wait_asynccnt 0x0" ::: "memory")
#define ASYNC_WAIT_2() asm volatile("s_wait_asynccnt 0x2" ::: "memory")
#else
#define ASYNC_WAIT_0()
#define ASYNC_WAIT_2()
#endif

#define ASG __attribute__((address_space(1)))
#define ASL __attribute__((address_space(3)))

union Frag { v16bf v; uint4 u[2]; };

// ---- WMMA fragment loaders ---------------------------------------------
// A: 16x32 bf16 (row-major src, leading dim ld). ISA layout:
//   lanes 0-15: M=lane, K=0..7 (V0-3), K=16..23 (V4-7)
//   lanes16-31: M=lane-16, K=8..15 (V0-3), K=24..31 (V4-7)
__device__ __forceinline__ v16bf load_a_frag(const bf16* __restrict__ src, int ld,
                                             int row0, int k0, int lane) {
  int r  = lane & 15;
  int ks = (lane >> 4) << 3;            // 0 or 8
  const bf16* p = src + (size_t)(row0 + r) * ld + k0 + ks;
  Frag f;
  f.u[0] = *(const uint4*)(p);
  f.u[1] = *(const uint4*)(p + 16);
  return f.v;
}

// B: 32x16 bf16 where B[k][n] = W[n][k], W row-major [N x K], ld = K.
//   lanes 0-15: N=lane, K=k0..k0+7 (V0-3), K=k0+16..k0+23 (V4-7)
//   lanes16-31: N=lane-16, K=k0+8..k0+15 (V0-3), K=k0+24..k0+31 (V4-7)
__device__ __forceinline__ v16bf load_b_frag(const bf16* __restrict__ w, int ld,
                                             int n0, int k0, int lane) {
  int n  = lane & 15;
  int kh = (lane >> 4) << 3;            // 0 or 8
  const bf16* p = w + (size_t)(n0 + n) * ld + k0 + kh;
  Frag f;
  f.u[0] = *(const uint4*)(p);
  f.u[1] = *(const uint4*)(p + 16);
  return f.v;
}

__device__ __forceinline__ v8f wmma_bf16(v16bf a, v16bf b, v8f c) {
  return __builtin_amdgcn_wmma_f32_16x16x32_bf16(false, a, false, b, (short)0, c,
                                                 false, false);
}

// ---- grid-wide barrier (8 resident WGs) ---------------------------------
__device__ __forceinline__ void gbar(unsigned* cnt, unsigned* gen) {
  __syncthreads();
  if (threadIdx.x == 0) {
    unsigned g = __hip_atomic_load(gen, __ATOMIC_ACQUIRE, __HIP_MEMORY_SCOPE_AGENT);
    unsigned a = __hip_atomic_fetch_add(cnt, 1u, __ATOMIC_ACQ_REL, __HIP_MEMORY_SCOPE_AGENT);
    if (a == NWG_SCAN - 1) {
      __hip_atomic_store(cnt, 0u, __ATOMIC_RELAXED, __HIP_MEMORY_SCOPE_AGENT);
      __hip_atomic_store(gen, g + 1u, __ATOMIC_RELEASE, __HIP_MEMORY_SCOPE_AGENT);
    } else {
      while (__hip_atomic_load(gen, __ATOMIC_ACQUIRE, __HIP_MEMORY_SCOPE_AGENT) == g)
        __builtin_amdgcn_s_sleep(2);
    }
  }
  __syncthreads();
}

// ---- utility kernels ----------------------------------------------------
__global__ void cvt_bf16_kernel(const float* __restrict__ s, bf16* __restrict__ d, int n) {
  for (int i = blockIdx.x * blockDim.x + threadIdx.x; i < n; i += gridDim.x * blockDim.x)
    d[i] = (bf16)s[i];
}

__global__ void zero_ctrl_kernel(unsigned* c) {
  if (threadIdx.x < 8) c[threadIdx.x] = 0u;
}

// ---- LTC scan: persistent kernel, 8 WGs x 256 threads -------------------
__global__ __launch_bounds__(256, 1) void scan_kernel(
    const bf16* __restrict__ xbf, const bf16* __restrict__ w0bf, const float* __restrict__ bin0,
    const bf16* __restrict__ wibf, const float* __restrict__ bi,
    const bf16* __restrict__ whbf, const float* __restrict__ bh,
    const bf16* __restrict__ wobf, const float* __restrict__ bo,
    const float* __restrict__ lng, const float* __restrict__ lnb,
    bf16* __restrict__ hbf, float* __restrict__ hsf, bf16* __restrict__ hsb,
    float* __restrict__ tmpf, bf16* __restrict__ seqbf, float* __restrict__ pooled_seq,
    unsigned* __restrict__ cnt, unsigned* __restrict__ gen)
{
  const int wg   = blockIdx.x;
  const int tid  = threadIdx.x;
  const int wave = tid >> 5;
  const int lane = tid & 31;
  const int mi   = wave & 3;                          // M tile (rows mi*16..)
  const int n0   = wg * 64 + ((wave >> 2) * 2) * 16;  // first of 2 N tiles
  const int lr   = lane & 15;
  const int lh   = lane >> 4;
  const int brow = wg * 8 + wave;                     // LayerNorm row ownership

  for (int i = wg * 256 + tid; i < 2 * LLAYERS * BB * HH; i += NWG_SCAN * 256)
    hsb[i] = (bf16)0.0f;
  for (int i = wg * 256 + tid; i < LLAYERS * BB * HH; i += NWG_SCAN * 256)
    hsf[i] = 0.0f;
  gbar(cnt, gen);

  float pacc[16];
#pragma unroll
  for (int i = 0; i < 16; ++i) pacc[i] = 0.0f;

  int par = 0;
  for (int t = 0; t < SS; ++t) {
    // ---- stage 0: h = x_t @ Win0^T + bin0 (K = 64) ----
    {
      v8f acc[2];
#pragma unroll
      for (int j = 0; j < 2; ++j) {
        float bv = bin0[n0 + j * 16 + lr];
#pragma unroll
        for (int r = 0; r < 8; ++r) acc[j][r] = bv;
      }
      for (int kt = 0; kt < FF; kt += 32) {
        v16bf a = load_a_frag(xbf + (size_t)t * FF, SS * FF, mi * 16, kt, lane);
#pragma unroll
        for (int j = 0; j < 2; ++j) {
          v16bf b = load_b_frag(w0bf, FF, n0 + j * 16, kt, lane);
          acc[j] = wmma_bf16(a, b, acc[j]);
        }
      }
#pragma unroll
      for (int j = 0; j < 2; ++j)
#pragma unroll
        for (int r = 0; r < 8; ++r) {
          int m = mi * 16 + lh * 8 + r;
          int n = n0 + j * 16 + lr;
          hbf[(size_t)m * HH + n] = (bf16)acc[j][r];
        }
    }
    gbar(cnt, gen);

    for (int l = 0; l < LLAYERS; ++l) {
      const bf16* hs_old = hsb + (size_t)par * LLAYERS * BB * HH + (size_t)l * BB * HH;
      bf16* hs_new = hsb + (size_t)(1 - par) * LLAYERS * BB * HH + (size_t)l * BB * HH;
      float* hsfl = hsf + (size_t)l * BB * HH;

      // ---- comb = h@Wi^T + hs@Wh^T + bi + bh ; liquid update ----
      {
        v8f acc[2];
#pragma unroll
        for (int j = 0; j < 2; ++j) {
          float bv = bi[l * HH + n0 + j * 16 + lr] + bh[l * HH + n0 + j * 16 + lr];
#pragma unroll
          for (int r = 0; r < 8; ++r) acc[j][r] = bv;
        }
        const bf16* wi = wibf + (size_t)l * HH * HH;
        const bf16* wh = whbf + (size_t)l * HH * HH;
        for (int kt = 0; kt < HH; kt += 32) {
          __builtin_prefetch(wi + (size_t)(n0 + lr) * HH + kt + 128, 0, 1);
          v16bf a1 = load_a_frag(hbf, HH, mi * 16, kt, lane);
          v16bf a2 = load_a_frag(hs_old, HH, mi * 16, kt, lane);
#pragma unroll
          for (int j = 0; j < 2; ++j) {
            v16bf b1 = load_b_frag(wi, HH, n0 + j * 16, kt, lane);
            v16bf b2 = load_b_frag(wh, HH, n0 + j * 16, kt, lane);
            acc[j] = wmma_bf16(a1, b1, acc[j]);
            acc[j] = wmma_bf16(a2, b2, acc[j]);
          }
        }
#pragma unroll
        for (int j = 0; j < 2; ++j)
#pragma unroll
          for (int r = 0; r < 8; ++r) {
            int m = mi * 16 + lh * 8 + r;
            int n = n0 + j * 16 + lr;
            size_t idx = (size_t)m * HH + n;
            float old = hsfl[idx];
            float nh  = old + ALPHA_C * (acc[j][r] - old);
            hsfl[idx] = nh;
            hs_new[idx] = (bf16)nh;
          }
      }
      gbar(cnt, gen);

      // ---- tmp = nh @ Wo^T + bo ----
      {
        v8f acc[2];
#pragma unroll
        for (int j = 0; j < 2; ++j) {
          float bv = bo[l * HH + n0 + j * 16 + lr];
#pragma unroll
          for (int r = 0; r < 8; ++r) acc[j][r] = bv;
        }
        const bf16* wo = wobf + (size_t)l * HH * HH;
        for (int kt = 0; kt < HH; kt += 32) {
          v16bf a = load_a_frag(hs_new, HH, mi * 16, kt, lane);
#pragma unroll
          for (int j = 0; j < 2; ++j) {
            v16bf b = load_b_frag(wo, HH, n0 + j * 16, kt, lane);
            acc[j] = wmma_bf16(a, b, acc[j]);
          }
        }
#pragma unroll
        for (int j = 0; j < 2; ++j)
#pragma unroll
          for (int r = 0; r < 8; ++r) {
            int m = mi * 16 + lh * 8 + r;
            int n = n0 + j * 16 + lr;
            tmpf[(size_t)m * HH + n] = acc[j][r];
          }
      }
      gbar(cnt, gen);

      // ---- LayerNorm over H=512 (wave owns row brow) ----
      {
        const float* row = tmpf + (size_t)brow * HH;
        float v[16];
        float s = 0.0f, sq = 0.0f;
#pragma unroll
        for (int i = 0; i < 16; i += 4) {
          float4 f4 = *(const float4*)(row + lane * 16 + i);
          v[i] = f4.x; v[i + 1] = f4.y; v[i + 2] = f4.z; v[i + 3] = f4.w;
        }
#pragma unroll
        for (int i = 0; i < 16; ++i) { s += v[i]; sq += v[i] * v[i]; }
        for (int d = 1; d < 32; d <<= 1) { s += __shfl_xor(s, d); sq += __shfl_xor(sq, d); }
        float mean = s * (1.0f / HH);
        float var  = sq * (1.0f / HH) - mean * mean;
        float rstd = rsqrtf(var + EPS_LN);
#pragma unroll
        for (int i = 0; i < 16; ++i) {
          int c = lane * 16 + i;
          float o = (v[i] - mean) * rstd * lng[l * HH + c] + lnb[l * HH + c];
          hbf[(size_t)brow * HH + c] = (bf16)o;
          if (l == LLAYERS - 1) {
            seqbf[((size_t)brow * SS + t) * HH + c] = (bf16)o;
            pacc[i] += o;
          }
        }
      }
      gbar(cnt, gen);
    }
    par ^= 1;
  }
#pragma unroll
  for (int i = 0; i < 16; ++i)
    pooled_seq[(size_t)brow * HH + lane * 16 + i] = pacc[i];
}

// ---- QKV projection GEMM: (B*S x 1536), K=512 ---------------------------
__global__ __launch_bounds__(256, 1) void qkv_kernel(
    const bf16* __restrict__ seqbf, const bf16* __restrict__ ipw, const float* __restrict__ ipb,
    bf16* __restrict__ qb, bf16* __restrict__ kb, bf16* __restrict__ vtb)
{
  const int tid = threadIdx.x, wave = tid >> 5, lane = tid & 31;
  const int mi = wave & 3, njb = (wave >> 2) * 2;
  const int lr = lane & 15, lh = lane >> 4;
  const int m0 = blockIdx.y * 64 + mi * 16;
  const int n0 = blockIdx.x * 64 + njb * 16;

  v8f acc[2];
#pragma unroll
  for (int j = 0; j < 2; ++j) {
    float bv = ipb[n0 + j * 16 + lr];
#pragma unroll
    for (int r = 0; r < 8; ++r) acc[j][r] = bv;
  }
  for (int kt = 0; kt < HH; kt += 32) {
    __builtin_prefetch(ipw + (size_t)(n0 + lr) * HH + kt + 128, 0, 1);
    v16bf a = load_a_frag(seqbf, HH, m0, kt, lane);
#pragma unroll
    for (int j = 0; j < 2; ++j) {
      v16bf b = load_b_frag(ipw, HH, n0 + j * 16, kt, lane);
      acc[j] = wmma_bf16(a, b, acc[j]);
    }
  }
#pragma unroll
  for (int j = 0; j < 2; ++j)
#pragma unroll
    for (int r = 0; r < 8; ++r) {
      int m = m0 + lh * 8 + r;
      int n = n0 + j * 16 + lr;
      int bidx = m >> 9, s = m & 511;
      int sel = n >> 9, hd = (n >> 6) & 7, d = n & 63;
      float v = acc[j][r];
      size_t bh = (size_t)(bidx * NHH + hd);
      if (sel == 0)      qb[(bh * SS + s) * DHH + d] = (bf16)(v * 0.125f);
      else if (sel == 1) kb[(bh * SS + s) * DHH + d] = (bf16)v;
      else               vtb[(bh * DHH + d) * SS + s] = (bf16)v;   // v transposed
    }
}

// ---- flash attention with double-buffered async K/V staging in LDS ------
__global__ __launch_bounds__(256, 1) void attn_kernel(
    const bf16* __restrict__ qb, const bf16* __restrict__ kb, const bf16* __restrict__ vtb,
    float* __restrict__ part)
{
  __shared__ bf16 kbuf[2][32 * DHH];   // [buf][key_local][d]   4KB each
  __shared__ bf16 vbuf[2][DHH * 32];   // [buf][d][s_local]     4KB each
  __shared__ bf16 plds[8][16 * 32];    // per-wave prob tiles   8KB
  const int tid = threadIdx.x, wave = tid >> 5, lane = tid & 31;
  const int lr = lane & 15, lh = lane >> 4;
  const int bh = blockIdx.x >> 2, mblk = blockIdx.x & 3;
  const bf16* q  = qb  + (size_t)bh * SS * DHH;
  const bf16* k  = kb  + (size_t)bh * SS * DHH;
  const bf16* vt = vtb + (size_t)bh * DHH * SS;
  const int q0 = mblk * 128 + wave * 16;

  v8f O[4];
  float rmax[8], rsum[8];
#pragma unroll
  for (int t4 = 0; t4 < 4; ++t4)
#pragma unroll
    for (int r = 0; r < 8; ++r) O[t4][r] = 0.0f;
#pragma unroll
  for (int r = 0; r < 8; ++r) { rmax[r] = -1e30f; rsum[r] = 0.0f; }

  v16bf aq0 = load_a_frag(q, DHH, q0, 0, lane);
  v16bf aq1 = load_a_frag(q, DHH, q0, 32, lane);

  // per-thread staging addresses: K panel 4KB (16B/thread), V panel 4KB
  const int vrow = tid >> 2, vcol = (tid & 3) * 16;  // bytes within 64B row

  auto stage = [&](int blk, int bufi) {
    int key0 = blk * 32;
    const char* gk = (const char*)(k + (size_t)key0 * DHH) + tid * 16;
    char* lk = (char*)&kbuf[bufi][0] + tid * 16;
    const char* gv = (const char*)(vt + (size_t)vrow * SS + key0) + vcol;
    char* lv = (char*)&vbuf[bufi][0] + vrow * 64 + vcol;
#if HAVE_ASYNC_LDS
    __builtin_amdgcn_global_load_async_to_lds_b128((ASG v4i*)gk, (ASL v4i*)lk, 0, 0);
    __builtin_amdgcn_global_load_async_to_lds_b128((ASG v4i*)gv, (ASL v4i*)lv, 0, 0);
#else
    *(uint4*)lv = *(const uint4*)gv;
    *(uint4*)lk = *(const uint4*)gk;
#endif
  };

  const int NB = SS / 32;
  int cur = 0;
  stage(0, 0);

  for (int kt = 0; kt < NB; ++kt) {
    __syncthreads();                       // previous reads of buf[cur^1] done
    if (kt + 1 < NB) { stage(kt + 1, cur ^ 1); ASYNC_WAIT_2(); }
    else             { ASYNC_WAIT_0(); }
    __syncthreads();                       // block kt fully staged in buf[cur]

    v8f s0, s1;
#pragma unroll
    for (int r = 0; r < 8; ++r) { s0[r] = 0.0f; s1[r] = 0.0f; }
    const bf16* kl = &kbuf[cur][0];
    s0 = wmma_bf16(aq0, load_b_frag(kl, DHH, 0,  0,  lane), s0);
    s0 = wmma_bf16(aq1, load_b_frag(kl, DHH, 0,  32, lane), s0);
    s1 = wmma_bf16(aq0, load_b_frag(kl, DHH, 16, 0,  lane), s1);
    s1 = wmma_bf16(aq1, load_b_frag(kl, DHH, 16, 32, lane), s1);

#pragma unroll
    for (int r = 0; r < 8; ++r) {
      float mr = fmaxf(s0[r], s1[r]);
      for (int d = 1; d < 16; d <<= 1) mr = fmaxf(mr, __shfl_xor(mr, d));
      float nm = fmaxf(rmax[r], mr);
      float sc = __expf(rmax[r] - nm);
      rmax[r] = nm;
      float p0 = __expf(s0[r] - nm);
      float p1 = __expf(s1[r] - nm);
      float ps = p0 + p1;
      for (int d = 1; d < 16; d <<= 1) ps += __shfl_xor(ps, d);
      rsum[r] = rsum[r] * sc + ps;
#pragma unroll
      for (int t4 = 0; t4 < 4; ++t4) O[t4][r] *= sc;
      int rowi = lh * 8 + r;
      plds[wave][rowi * 32 + lr]      = (bf16)p0;
      plds[wave][rowi * 32 + 16 + lr] = (bf16)p1;
    }
    __syncthreads();
    v16bf ap = load_a_frag(&plds[wave][0], 32, 0, 0, lane);
    const bf16* vl = &vbuf[cur][0];
#pragma unroll
    for (int t4 = 0; t4 < 4; ++t4)
      O[t4] = wmma_bf16(ap, load_b_frag(vl, 32, t4 * 16, 0, lane), O[t4]);
    cur ^= 1;
  }

  float inv[8];
#pragma unroll
  for (int r = 0; r < 8; ++r) inv[r] = 1.0f / rsum[r];
  // column sums over this wave's 16 query rows -> pooled partials
#pragma unroll
  for (int t4 = 0; t4 < 4; ++t4) {
    float cs = 0.0f;
#pragma unroll
    for (int r = 0; r < 8; ++r) cs += O[t4][r] * inv[r];
    cs += __shfl_xor(cs, 16);
    if (lh == 0)
      part[(((size_t)bh * 32) + mblk * 8 + wave) * DHH + t4 * 16 + lr] = cs;
  }
}

// ---- reduce attention partials -> sum_t(attn_out) [B x H] ---------------
__global__ void reduce_ao_kernel(const float* __restrict__ part, float* __restrict__ pA) {
  int idx = blockIdx.x * blockDim.x + threadIdx.x;
  if (idx >= BB * HH) return;
  int b = idx >> 9, c = idx & 511;
  int hd = c >> 6, d = c & 63;
  const float* p = part + (size_t)(b * NHH + hd) * 32 * DHH + d;
  float s = 0.0f;
  for (int i = 0; i < 32; ++i) s += p[(size_t)i * DHH];
  pA[idx] = s;
}

// ---- pooled + output projection + MLP heads (single WG) -----------------
__global__ __launch_bounds__(256, 1) void heads_kernel(
    const float* __restrict__ pooled_seq, const float* __restrict__ pA,
    const float* __restrict__ outw, const float* __restrict__ outb,
    const float* __restrict__ Wd1, const float* __restrict__ bd1,
    const float* __restrict__ Wd2, const float* __restrict__ bd2,
    const float* __restrict__ Wd3, const float* __restrict__ bd3,
    const float* __restrict__ Wq1, const float* __restrict__ bq1,
    const float* __restrict__ Wq2, const float* __restrict__ bq2,
    const float* __restrict__ Wq3, const float* __restrict__ bq3,
    const float* __restrict__ Wc1, const float* __restrict__ bc1,
    const float* __restrict__ Wc2, const float* __restrict__ bc2,
    float* __restrict__ pooled, float* __restrict__ h1, float* __restrict__ h2,
    float* __restrict__ out)
{
  const int tid = threadIdx.x;
  const float invS = 1.0f / (float)SS;

  for (int idx = tid; idx < BB * HH; idx += 256) {
    int b = idx >> 9, n = idx & 511;
    const float* pa = pA + ((size_t)b << 9);
    const float* w  = outw + (size_t)n * HH;
    float s = 0.0f;
    for (int kk = 0; kk < HH; ++kk) s += pa[kk] * w[kk];
    pooled[idx] = pooled_seq[idx] * invS + s * invS + outb[n];
  }
  __threadfence_block();
  __syncthreads();

  for (int idx = tid; idx < 3 * BB * 256; idx += 256) {
    int sel = idx / (BB * 256);
    int rem = idx - sel * (BB * 256);
    int b = rem >> 8, j = rem & 255;
    const float* W = (sel == 0) ? Wd1 : (sel == 1) ? Wq1 : Wc1;
    const float* Bv = (sel == 0) ? bd1 : (sel == 1) ? bq1 : bc1;
    const float* in = pooled + ((size_t)b << 9);
    const float* wr = W + (size_t)j * HH;
    float s = Bv[j];
    for (int kk = 0; kk < HH; ++kk) s += in[kk] * wr[kk];
    h1[idx] = fmaxf(s, 0.0f);
  }
  __threadfence_block();
  __syncthreads();

  for (int idx = tid; idx < 2 * BB * 128; idx += 256) {
    int sel = idx / (BB * 128);
    int rem = idx - sel * (BB * 128);
    int b = rem >> 7, j = rem & 127;
    const float* W = (sel == 0) ? Wd2 : Wq2;
    const float* Bv = (sel == 0) ? bd2 : bq2;
    const float* in = h1 + (size_t)sel * BB * 256 + (size_t)b * 256;
    const float* wr = W + (size_t)j * 256;
    float s = Bv[j];
    for (int kk = 0; kk < 256; ++kk) s += in[kk] * wr[kk];
    h2[idx] = fmaxf(s, 0.0f);
  }
  __threadfence_block();
  __syncthreads();

  if (tid < 128) {
    int b = tid >> 1, c = tid & 1;
    const float* in = h2 + (size_t)b * 128;
    const float* wr = Wd3 + (size_t)c * 128;
    float s = bd3[c];
    for (int kk = 0; kk < 128; ++kk) s += in[kk] * wr[kk];
    out[tid] = s;
  } else if (tid < 192) {
    int b = tid - 128;
    const float* in = h2 + (size_t)BB * 128 + (size_t)b * 128;
    float s = bq3[0];
    for (int kk = 0; kk < 128; ++kk) s += in[kk] * Wq3[kk];
    out[tid] = s;
  } else {
    int b = tid - 192;
    const float* in = h1 + (size_t)2 * BB * 256 + (size_t)b * 256;
    float s = bc2[0];
    for (int kk = 0; kk < 256; ++kk) s += in[kk] * Wc2[kk];
    out[tid] = 1.0f / (1.0f + __expf(-s));
  }
}

// ---- host launch --------------------------------------------------------
extern "C" void kernel_launch(void* const* d_in, const int* in_sizes, int n_in,
                              void* d_out, int out_size, void* d_ws, size_t ws_size,
                              hipStream_t stream) {
  (void)in_sizes; (void)n_in; (void)out_size; (void)ws_size;
  const float* x     = (const float*)d_in[0];
  const float* Win0  = (const float*)d_in[1];
  const float* bin0  = (const float*)d_in[2];
  const float* Wi    = (const float*)d_in[3];
  const float* bi    = (const float*)d_in[4];
  const float* Wh    = (const float*)d_in[5];
  const float* bh    = (const float*)d_in[6];
  const float* Wo    = (const float*)d_in[7];
  const float* bo    = (const float*)d_in[8];
  const float* ln_g  = (const float*)d_in[9];
  const float* ln_b  = (const float*)d_in[10];
  const float* ipw   = (const float*)d_in[11];
  const float* ipb   = (const float*)d_in[12];
  const float* outw  = (const float*)d_in[13];
  const float* outb  = (const float*)d_in[14];
  const float* Wd1 = (const float*)d_in[15]; const float* bd1 = (const float*)d_in[16];
  const float* Wd2 = (const float*)d_in[17]; const float* bd2 = (const float*)d_in[18];
  const float* Wd3 = (const float*)d_in[19]; const float* bd3 = (const float*)d_in[20];
  const float* Wq1 = (const float*)d_in[21]; const float* bq1 = (const float*)d_in[22];
  const float* Wq2 = (const float*)d_in[23]; const float* bq2 = (const float*)d_in[24];
  const float* Wq3 = (const float*)d_in[25]; const float* bq3 = (const float*)d_in[26];
  const float* Wc1 = (const float*)d_in[27]; const float* bc1 = (const float*)d_in[28];
  const float* Wc2 = (const float*)d_in[29]; const float* bc2 = (const float*)d_in[30];

  char* ws = (char*)d_ws;
  size_t o = 0;
  auto alloc = [&](size_t bytes) {
    size_t r = (o + 255) & ~(size_t)255;
    o = r + bytes;
    return r;
  };
  size_t oXBF   = alloc((size_t)BB * SS * FF * 2);
  size_t oW0BF  = alloc((size_t)HH * FF * 2);
  size_t oWIBF  = alloc((size_t)LLAYERS * HH * HH * 2);
  size_t oWHBF  = alloc((size_t)LLAYERS * HH * HH * 2);
  size_t oWOBF  = alloc((size_t)LLAYERS * HH * HH * 2);
  size_t oIPWBF = alloc((size_t)3 * HH * HH * 2);
  size_t oHBF   = alloc((size_t)BB * HH * 2);
  size_t oHSF   = alloc((size_t)LLAYERS * BB * HH * 4);
  size_t oHSB   = alloc((size_t)2 * LLAYERS * BB * HH * 2);
  size_t oTMPF  = alloc((size_t)BB * HH * 4);
  size_t oSEQBF = alloc((size_t)BB * SS * HH * 2);
  size_t oPSEQ  = alloc((size_t)BB * HH * 4);
  size_t oQB    = alloc((size_t)BB * NHH * SS * DHH * 2);
  size_t oKB    = alloc((size_t)BB * NHH * SS * DHH * 2);
  size_t oVTB   = alloc((size_t)BB * NHH * DHH * SS * 2);
  size_t oPART  = alloc((size_t)BB * NHH * 32 * DHH * 4);
  size_t oPA    = alloc((size_t)BB * HH * 4);
  size_t oPOOL  = alloc((size_t)BB * HH * 4);
  size_t oH1    = alloc((size_t)3 * BB * 256 * 4);
  size_t oH2    = alloc((size_t)2 * BB * 128 * 4);
  size_t oCTRL  = alloc(256);

  bf16* xbf   = (bf16*)(ws + oXBF);
  bf16* w0bf  = (bf16*)(ws + oW0BF);
  bf16* wibf  = (bf16*)(ws + oWIBF);
  bf16* whbf  = (bf16*)(ws + oWHBF);
  bf16* wobf  = (bf16*)(ws + oWOBF);
  bf16* ipwbf = (bf16*)(ws + oIPWBF);
  bf16* hbf   = (bf16*)(ws + oHBF);
  float* hsf  = (float*)(ws + oHSF);
  bf16* hsb   = (bf16*)(ws + oHSB);
  float* tmpf = (float*)(ws + oTMPF);
  bf16* seqbf = (bf16*)(ws + oSEQBF);
  float* pseq = (float*)(ws + oPSEQ);
  bf16* qb    = (bf16*)(ws + oQB);
  bf16* kb    = (bf16*)(ws + oKB);
  bf16* vtb   = (bf16*)(ws + oVTB);
  float* part = (float*)(ws + oPART);
  float* pA   = (float*)(ws + oPA);
  float* pooled = (float*)(ws + oPOOL);
  float* h1   = (float*)(ws + oH1);
  float* h2   = (float*)(ws + oH2);
  unsigned* ctrl = (unsigned*)(ws + oCTRL);

  zero_ctrl_kernel<<<1, 32, 0, stream>>>(ctrl);
  cvt_bf16_kernel<<<512, 256, 0, stream>>>(x,    xbf,   BB * SS * FF);
  cvt_bf16_kernel<<<64,  256, 0, stream>>>(Win0, w0bf,  HH * FF);
  cvt_bf16_kernel<<<512, 256, 0, stream>>>(Wi,   wibf,  LLAYERS * HH * HH);
  cvt_bf16_kernel<<<512, 256, 0, stream>>>(Wh,   whbf,  LLAYERS * HH * HH);
  cvt_bf16_kernel<<<512, 256, 0, stream>>>(Wo,   wobf,  LLAYERS * HH * HH);
  cvt_bf16_kernel<<<512, 256, 0, stream>>>(ipw,  ipwbf, 3 * HH * HH);

  scan_kernel<<<NWG_SCAN, 256, 0, stream>>>(
      xbf, w0bf, bin0, wibf, bi, whbf, bh, wobf, bo, ln_g, ln_b,
      hbf, hsf, hsb, tmpf, seqbf, pseq, ctrl, ctrl + 1);

  qkv_kernel<<<dim3(3 * HH / 64, BB * SS / 64), 256, 0, stream>>>(
      seqbf, ipwbf, ipb, qb, kb, vtb);

  attn_kernel<<<BB * NHH * 4, 256, 0, stream>>>(qb, kb, vtb, part);

  reduce_ao_kernel<<<(BB * HH + 255) / 256, 256, 0, stream>>>(part, pA);

  heads_kernel<<<1, 256, 0, stream>>>(
      pseq, pA, outw, outb,
      Wd1, bd1, Wd2, bd2, Wd3, bd3,
      Wq1, bq1, Wq2, bq2, Wq3, bq3,
      Wc1, bc1, Wc2, bc2,
      pooled, h1, h2, (float*)d_out);
}
